// GAT_17600775979729
// MI455X (gfx1250) — matrix operand order repaired and checked by reading, not compiled, not run on hardware
//
#include <hip/hip_runtime.h>
#include <hip/hip_bf16.h>
#include <math.h>

// ---------------------------------------------------------------------------
// GAT on MI455X (gfx1250): bf16 WMMA GEMMs (TDM-staged B panel in LDS)
// + atomic segment ops for the graph part.
// ---------------------------------------------------------------------------

typedef __attribute__((ext_vector_type(16))) __bf16 v16bf;
typedef __attribute__((ext_vector_type(8)))  __bf16 v8bf;
typedef __attribute__((ext_vector_type(8)))  float  v8f;
typedef __attribute__((ext_vector_type(4)))  unsigned int v4u;
typedef __attribute__((ext_vector_type(8)))  int v8i;
typedef __attribute__((ext_vector_type(4)))  int v4i;

#define NEG_SLOPE 0.2f

// ---------------- float atomic max via int punning (init with -inf) --------
__device__ __forceinline__ void atomicMaxF(float* addr, float value) {
    if (value >= 0.0f) {
        atomicMax(reinterpret_cast<int*>(addr), __float_as_int(value));
    } else {
        atomicMin(reinterpret_cast<unsigned int*>(addr), __float_as_uint(value));
    }
}

__device__ __forceinline__ void atomicAddF(float* addr, float value) {
    unsafeAtomicAdd(addr, value);   // hardware global_atomic_add_f32
}

// ---------------- generic fill ---------------------------------------------
__global__ void gat_fill_f32(float* __restrict__ p, float v, size_t n) {
    size_t i = (size_t)blockIdx.x * blockDim.x + threadIdx.x;
    size_t stride = (size_t)gridDim.x * blockDim.x;
    for (; i < n; i += stride) p[i] = v;
}

// ---------------- f32 -> bf16 with row padding -----------------------------
__global__ void gat_cvt_bf16_pad(const float* __restrict__ src, __bf16* __restrict__ dst,
                                 int rows, int cols, int padRows) {
    size_t idx = (size_t)blockIdx.x * blockDim.x + threadIdx.x;
    size_t total = (size_t)padRows * cols;
    if (idx >= total) return;
    size_t r = idx / (size_t)cols;
    float v = (r < (size_t)rows) ? src[idx] : 0.0f;
    dst[idx] = (__bf16)v;
}

// ---------------- pack W[K x Nn] (row major f32) into WMMA-B layout --------
// Wp[(kb*Nn + n)*32 + j] = W[(kb*32 + j)*Nn + n]
__global__ void gat_pack_w(const float* __restrict__ W, __bf16* __restrict__ Wp,
                           int K, int Nn) {
    int tid = blockIdx.x * blockDim.x + threadIdx.x;
    if (tid >= K * Nn) return;
    int j  = tid & 31;
    int q  = tid >> 5;
    int n  = q % Nn;
    int kb = q / Nn;
    Wp[tid] = (__bf16)W[(size_t)(kb * 32 + j) * Nn + n];
}

// ---------------- WMMA GEMM: C[M x Nn] = A[M x K] * B[K x Nn] --------------
// A: bf16 row-major (lda = K), Bp: packed (see gat_pack_w), C: f32 row-major.
// block = 256 (8 waves), each wave computes a 16x64 tile; grid (M/128, Nn/64).
// The 64-column B panel (all K) is staged once per block into LDS with the
// Tensor Data Mover (TENSORcnt + s_wait_tensorcnt), then all 8 waves read B
// fragments from LDS; A fragments are register double-buffered.
__global__ __launch_bounds__(256)
void gat_gemm_wmma(const __bf16* __restrict__ A, const __bf16* __restrict__ Bp,
                   float* __restrict__ C, int M, int K, int Nn) {
    extern __shared__ char gat_lds[];     // (K/32) * 4096 bytes

    const int lane  = threadIdx.x & 31;
    const int wave  = threadIdx.x >> 5;
    const int m0    = blockIdx.x * 128 + wave * 16;
    const int n0    = blockIdx.y * 64;
    const int mrow  = m0 + (lane & 15);
    const int khalf = lane >> 4;          // 0: K 0..15, 1: K 16..31
    const int kb_end = K >> 5;

    // ---- TDM: stage packed B panel [kb_end rows x 4096 B] into LDS --------
    if (wave == 0) {
        const unsigned long long ga =
            (unsigned long long)(Bp + (size_t)n0 * 32);           // tile start
        const unsigned int lds_addr = (unsigned int)(size_t)gat_lds;
        // D# group 0: count=1 | lds_addr | global_addr[56:0] | type=2
        v4u g0;
        g0.x = 1u;                                                // count=1, user mode
        g0.y = lds_addr;
        g0.z = (unsigned int)(ga & 0xFFFFFFFFu);
        g0.w = ((unsigned int)(ga >> 32) & 0x01FFFFFFu) | (2u << 30);
        // D# group 1 (256b): data_size=3 (8B units); 2-D tile:
        //   tile_dim0 = 512 units (4096 B row), tile_dim1 = kb_end rows
        //   tensor_dim0 = 512, tensor_dim1 = kb_end
        //   tensor_dim0_stride = Nn*8 units (= Nn*64 B = packed k-slab pitch)
        const unsigned int td0 = 512u, td1 = (unsigned int)kb_end;
        const unsigned int tile0 = 512u, tile1 = (unsigned int)kb_end;
        const unsigned long long st0 = (unsigned long long)Nn * 8ull;
        v8i g1;
        g1[0] = (int)(3u << 16);                                  // data_size=8B
        g1[1] = (int)((td0 & 0xFFFFu) << 16);                     // tensor_dim0 lo
        g1[2] = (int)((td0 >> 16) | ((td1 & 0xFFFFu) << 16));     // td0 hi | td1 lo
        g1[3] = (int)((td1 >> 16) | (tile0 << 16));               // td1 hi | tile_dim0
        g1[4] = (int)tile1;                                       // tile_dim1 (tile_dim2=0)
        g1[5] = (int)(unsigned int)(st0 & 0xFFFFFFFFull);         // stride0 lo
        g1[6] = (int)(unsigned int)(st0 >> 32);                   // stride0 hi | stride1 lo(=0)
        g1[7] = 0;                                                // stride1 hi
        v4i g2 = {0, 0, 0, 0};                                    // 2-D tensor: unused
        v4i g3 = {0, 0, 0, 0};
        v8i g4 = {0, 0, 0, 0, 0, 0, 0, 0};
        __builtin_amdgcn_tensor_load_to_lds(g0, g1, g2, g3, g4, 0);
        __builtin_amdgcn_s_wait_tensorcnt(0);
    }
    __syncthreads();

    // ---- main K loop: B from LDS, A double-buffered from global -----------
    const __bf16* abase = A + (size_t)mrow * K + khalf * 16;
    v8bf lo = *reinterpret_cast<const v8bf*>(abase);
    v8bf hi = *reinterpret_cast<const v8bf*>(abase + 8);

    v8f acc0 = {}, acc1 = {}, acc2 = {}, acc3 = {};
    const size_t bcol = ((size_t)(lane & 15) << 6) + ((size_t)khalf << 5);

    for (int kb = 0; kb < kb_end; ++kb) {
        v8bf lo_n = lo, hi_n = hi;
        if (kb + 1 < kb_end) {
            const __bf16* an = abase + (size_t)(kb + 1) * 32;
            lo_n = *reinterpret_cast<const v8bf*>(an);
            hi_n = *reinterpret_cast<const v8bf*>(an + 8);
        }

        v16bf af;
#pragma unroll
        for (int i = 0; i < 8; ++i) { af[i] = lo[i]; af[i + 8] = hi[i]; }

        const char* bs = gat_lds + ((size_t)kb << 12) + bcol;     // kb*4096 + col*64 + khalf*32
        v16bf b0 = *reinterpret_cast<const v16bf*>(bs);
        v16bf b1 = *reinterpret_cast<const v16bf*>(bs + 16 * 64);
        v16bf b2 = *reinterpret_cast<const v16bf*>(bs + 32 * 64);
        v16bf b3 = *reinterpret_cast<const v16bf*>(bs + 48 * 64);

        acc0 = __builtin_amdgcn_wmma_f32_16x16x32_bf16(false, af, false, b0, (short)0, acc0, false, false);
        acc1 = __builtin_amdgcn_wmma_f32_16x16x32_bf16(false, af, false, b1, (short)0, acc1, false, false);
        acc2 = __builtin_amdgcn_wmma_f32_16x16x32_bf16(false, af, false, b2, (short)0, acc2, false, false);
        acc3 = __builtin_amdgcn_wmma_f32_16x16x32_bf16(false, af, false, b3, (short)0, acc3, false, false);

        lo = lo_n; hi = hi_n;
    }

    // D layout: VGPR r -> row m0 + khalf*8 + r, column n0 + t*16 + (lane&15)
    const int srow = m0 + khalf * 8;
    const int scol = n0 + (lane & 15);
#pragma unroll
    for (int r = 0; r < 8; ++r) {
        float* cp = C + (size_t)(srow + r) * Nn + scol;
        cp[0]  = acc0[r];
        cp[16] = acc1[r];
        cp[32] = acc2[r];
        cp[48] = acc3[r];
    }
}

// ---------------- per-node attention scalars: s = h.a_s, d = h.a_d ---------
__global__ void gat_node_sd(const float* __restrict__ h,
                            const float* __restrict__ a_s, const float* __restrict__ a_d,
                            float* __restrict__ s, float* __restrict__ d,
                            int n_nodes, int F) {
    int gtid = blockIdx.x * blockDim.x + threadIdx.x;
    int node = gtid >> 5;
    int lane = threadIdx.x & 31;
    if (node >= n_nodes) return;
    float ss = 0.0f, dd = 0.0f;
    const float* hr = h + (size_t)node * F;
    for (int j = lane; j < F; j += 32) {
        float v = hr[j];
        ss += v * a_s[j];
        dd += v * a_d[j];
    }
    for (int o = 16; o > 0; o >>= 1) {
        ss += __shfl_down(ss, o, 32);
        dd += __shfl_down(dd, o, 32);
    }
    if (lane == 0) { s[node] = ss; d[node] = dd; }
}

// ---------------- edge pass 1: leaky-relu logit + segment max --------------
__global__ void gat_edge_logit_max(const int* __restrict__ ei, int E, int Ntot,
                                   const float* __restrict__ s, const float* __restrict__ d,
                                   float* __restrict__ elog, float* __restrict__ m) {
    int e = blockIdx.x * blockDim.x + threadIdx.x;
    if (e >= E + Ntot) return;
    int src = (e < E) ? ei[e]     : (e - E);
    int dst = (e < E) ? ei[E + e] : (e - E);
    float x = s[src] + d[dst];
    float lr = (x > 0.0f) ? x : NEG_SLOPE * x;
    elog[e] = lr;
    atomicMaxF(m + dst, lr);
}

// ---------------- edge pass 2: exp + segment denom -------------------------
__global__ void gat_edge_exp_denom(const int* __restrict__ ei, int E, int Ntot,
                                   const float* __restrict__ elog, const float* __restrict__ m,
                                   float* __restrict__ wexp, float* __restrict__ denom) {
    int e = blockIdx.x * blockDim.x + threadIdx.x;
    if (e >= E + Ntot) return;
    int dst = (e < E) ? ei[E + e] : (e - E);
    float mm = m[dst];
    if (!isfinite(mm)) mm = 0.0f;           // matches reference guard
    float w = __expf(elog[e] - mm);
    wexp[e] = w;
    atomicAddF(denom + dst, w);
}

// ---------------- edge pass 3: scatter alpha * h[src] into aggr[dst] -------
// one wave per edge, 16 features per lane (512 total)
__global__ __launch_bounds__(256)
void gat_edge_scatter(const int* __restrict__ ei, int E, int Ntot,
                      const float* __restrict__ wexp, const float* __restrict__ denom,
                      const float* __restrict__ h, float* __restrict__ aggr) {
    int gtid = blockIdx.x * blockDim.x + threadIdx.x;
    int e    = gtid >> 5;
    int lane = threadIdx.x & 31;
    if (e >= E + Ntot) return;
    int src = (e < E) ? ei[e]     : (e - E);
    int dst = (e < E) ? ei[E + e] : (e - E);
    float alpha = wexp[e] / denom[dst];
    const float4* hp = reinterpret_cast<const float4*>(h + (size_t)src * 512);
    float* ap = aggr + (size_t)dst * 512;
#pragma unroll
    for (int q = 0; q < 4; ++q) {
        int c4 = lane + q * 32;             // float4 index 0..127
        float4 v = hp[c4];
        float* a4 = ap + c4 * 4;
        atomicAddF(a4 + 0, alpha * v.x);
        atomicAddF(a4 + 1, alpha * v.y);
        atomicAddF(a4 + 2, alpha * v.z);
        atomicAddF(a4 + 3, alpha * v.w);
    }
}

// ---------------- epilogue: tanh(aggr + b), emit f32 + bf16 (padded) -------
__global__ void gat_epilogue(const float* __restrict__ aggr, const float* __restrict__ bias,
                             float* __restrict__ hfin, __bf16* __restrict__ xb,
                             int n_nodes, int padRows, int F) {
    size_t idx = (size_t)blockIdx.x * blockDim.x + threadIdx.x;
    size_t total = (size_t)padRows * F;
    if (idx >= total) return;
    int c = (int)(idx % (size_t)F);
    size_t r = idx / (size_t)F;
    float v = 0.0f;
    if (r < (size_t)n_nodes) v = tanhf(aggr[idx] + bias[c]);
    hfin[idx] = v;
    xb[idx] = (__bf16)v;
}

// ---------------- graph pooling: segment max + sum + count -----------------
__global__ void gat_pool(const float* __restrict__ hfin, const int* __restrict__ batch,
                         float* __restrict__ maxp, float* __restrict__ sump,
                         float* __restrict__ cnt, int n_nodes, int F) {
    int gtid = blockIdx.x * blockDim.x + threadIdx.x;
    int node = gtid >> 5;
    int lane = threadIdx.x & 31;
    if (node >= n_nodes) return;
    int g = batch[node];
    const float* hr = hfin + (size_t)node * F;
    for (int j = lane; j < F; j += 32) {
        float v = hr[j];
        atomicMaxF(maxp + (size_t)g * F + j, v);
        atomicAddF(sump + (size_t)g * F + j, v);
    }
    if (lane == 0) atomicAddF(cnt + g, 1.0f);
}

// ---------------- build z = [maxp | meanp] as padded bf16 ------------------
__global__ void gat_build_z(const float* __restrict__ maxp, const float* __restrict__ sump,
                            const float* __restrict__ cnt, __bf16* __restrict__ zb,
                            int G_, int Gpad) {
    size_t idx = (size_t)blockIdx.x * blockDim.x + threadIdx.x;
    size_t total = (size_t)Gpad * 1024;
    if (idx >= total) return;
    int c = (int)(idx & 1023);
    int g = (int)(idx >> 10);
    float v = 0.0f;
    if (g < G_) {
        if (c < 512) {
            v = maxp[(size_t)g * 512 + c];
            if (!isfinite(v)) v = 0.0f;
        } else {
            float cc = cnt[g];
            v = sump[(size_t)g * 512 + (c - 512)] / fmaxf(cc, 1.0f);
        }
    }
    zb[idx] = (__bf16)v;
}

// ---------------- lin1 activation: in-place tanh(zz + b) -------------------
__global__ void gat_lin_act(float* __restrict__ zz, const float* __restrict__ b,
                            int rows, int cols) {
    size_t idx = (size_t)blockIdx.x * blockDim.x + threadIdx.x;
    if (idx >= (size_t)rows * cols) return;
    int c = (int)(idx % (size_t)cols);
    zz[idx] = tanhf(zz[idx] + b[c]);
}

// ---------------- head: lin2 + log_softmax, one wave per graph -------------
__global__ void gat_head(const float* __restrict__ t, const float* __restrict__ W2,
                         const float* __restrict__ b2, float* __restrict__ out, int G_) {
    __shared__ float sl[16];
    int g = blockIdx.x;
    int lane = threadIdx.x;
    if (g >= G_) return;
    if (lane < 10) {
        float acc = b2[lane];
        const float* tr = t + (size_t)g * 512;
        for (int k = 0; k < 512; ++k) acc += tr[k] * W2[k * 10 + lane];
        sl[lane] = acc;
    }
    __syncthreads();
    if (lane < 10) {
        float mx = sl[0];
#pragma unroll
        for (int i = 1; i < 10; ++i) mx = fmaxf(mx, sl[i]);
        float se = 0.0f;
#pragma unroll
        for (int i = 0; i < 10; ++i) se += __expf(sl[i] - mx);
        out[(size_t)g * 10 + lane] = sl[lane] - mx - __logf(se);
    }
}

// ===========================================================================
extern "C" void kernel_launch(void* const* d_in, const int* in_sizes, int n_in,
                              void* d_out, int out_size, void* d_ws, size_t ws_size,
                              hipStream_t stream) {
    (void)in_sizes; (void)n_in; (void)out_size; (void)ws_size;

    constexpr int N_ = 100000, E_ = 400000, G_ = 1000, H_ = 512, FIN = 128;
    constexpr int Mpad = 100096;           // 782 * 128
    constexpr int Gpad = 1024;
    constexpr int Etot = E_ + N_;          // with self-loops

    const float* x     = (const float*)d_in[0];
    const int*   ei    = (const int*)d_in[1];
    const int*   batch = (const int*)d_in[2];
    const float* W[4]  = {(const float*)d_in[3],  (const float*)d_in[7],
                          (const float*)d_in[11], (const float*)d_in[15]};
    const float* As[4] = {(const float*)d_in[4],  (const float*)d_in[8],
                          (const float*)d_in[12], (const float*)d_in[16]};
    const float* Ad[4] = {(const float*)d_in[5],  (const float*)d_in[9],
                          (const float*)d_in[13], (const float*)d_in[17]};
    const float* Bb[4] = {(const float*)d_in[6],  (const float*)d_in[10],
                          (const float*)d_in[14], (const float*)d_in[18]};
    const float* lin1_W = (const float*)d_in[19];
    const float* lin1_b = (const float*)d_in[20];
    const float* lin2_W = (const float*)d_in[21];
    const float* lin2_b = (const float*)d_in[22];
    float* out = (float*)d_out;

    // ---- workspace carve-out (all offsets 256B aligned) ----
    char* ws = (char*)d_ws;
    size_t off = 0;
    auto take = [&](size_t bytes) -> char* {
        char* p = ws + off;
        off += (bytes + 255) & ~(size_t)255;
        return p;
    };
    float*  h     = (float*)take((size_t)Mpad * H_ * 4);     // GEMM out / tanh out
    float*  aggr  = (float*)take((size_t)Mpad * H_ * 4);     // message accumulator
    __bf16* Xbf   = (__bf16*)take((size_t)Mpad * H_ * 2);    // bf16 GEMM input
    __bf16* Wp    = (__bf16*)take((size_t)1024 * 512 * 2);   // packed weights
    float*  sarr  = (float*)take((size_t)N_ * 4);
    float*  darr  = (float*)take((size_t)N_ * 4);
    float*  mseg  = (float*)take((size_t)N_ * 4);
    float*  denom = (float*)take((size_t)N_ * 4);
    float*  elog  = (float*)take((size_t)Etot * 4);
    float*  wexp  = (float*)take((size_t)Etot * 4);
    float*  maxp  = (float*)take((size_t)G_ * H_ * 4);
    float*  sump  = (float*)take((size_t)G_ * H_ * 4);
    float*  cnt   = (float*)take((size_t)G_ * 4);
    __bf16* Zbf   = (__bf16*)take((size_t)Gpad * 1024 * 2);
    float*  zz    = (float*)take((size_t)Gpad * 512 * 4);

    auto cdiv = [](size_t a, size_t b) -> unsigned { return (unsigned)((a + b - 1) / b); };

    // ---- 4 GAT layers ----
    for (int l = 0; l < 4; ++l) {
        const int Kin = (l == 0) ? FIN : H_;
        const size_t ldsB = (size_t)(Kin >> 5) * 4096;   // staged B panel bytes

        gat_pack_w<<<cdiv((size_t)Kin * H_, 256), 256, 0, stream>>>(W[l], Wp, Kin, H_);
        if (l == 0)
            gat_cvt_bf16_pad<<<cdiv((size_t)Mpad * FIN, 256), 256, 0, stream>>>(
                x, Xbf, N_, FIN, Mpad);

        dim3 gg(Mpad / 128, H_ / 64);
        gat_gemm_wmma<<<gg, 256, ldsB, stream>>>(Xbf, Wp, h, Mpad, Kin, H_);

        gat_node_sd<<<cdiv((size_t)N_ * 32, 256), 256, 0, stream>>>(
            h, As[l], Ad[l], sarr, darr, N_, H_);

        gat_fill_f32<<<cdiv((size_t)N_, 256), 256, 0, stream>>>(mseg, -INFINITY, (size_t)N_);
        gat_fill_f32<<<cdiv((size_t)N_, 256), 256, 0, stream>>>(denom, 0.0f, (size_t)N_);
        gat_fill_f32<<<cdiv((size_t)N_ * H_, 256), 256, 0, stream>>>(
            aggr, 0.0f, (size_t)N_ * H_);

        gat_edge_logit_max<<<cdiv((size_t)Etot, 256), 256, 0, stream>>>(
            ei, E_, N_, sarr, darr, elog, mseg);
        gat_edge_exp_denom<<<cdiv((size_t)Etot, 256), 256, 0, stream>>>(
            ei, E_, N_, elog, mseg, wexp, denom);
        gat_edge_scatter<<<cdiv((size_t)Etot * 32, 256), 256, 0, stream>>>(
            ei, E_, N_, wexp, denom, h, aggr);

        gat_epilogue<<<cdiv((size_t)Mpad * H_, 256), 256, 0, stream>>>(
            aggr, Bb[l], h, Xbf, N_, Mpad, H_);
    }

    // ---- pooling ----
    gat_fill_f32<<<cdiv((size_t)G_ * H_, 256), 256, 0, stream>>>(maxp, -INFINITY, (size_t)G_ * H_);
    gat_fill_f32<<<cdiv((size_t)G_ * H_, 256), 256, 0, stream>>>(sump, 0.0f, (size_t)G_ * H_);
    gat_fill_f32<<<cdiv((size_t)G_, 256), 256, 0, stream>>>(cnt, 0.0f, (size_t)G_);
    gat_pool<<<cdiv((size_t)N_ * 32, 256), 256, 0, stream>>>(h, batch, maxp, sump, cnt, N_, H_);
    gat_build_z<<<cdiv((size_t)Gpad * 1024, 256), 256, 0, stream>>>(maxp, sump, cnt, Zbf, G_, Gpad);

    // ---- MLP head ----
    gat_pack_w<<<cdiv((size_t)1024 * 512, 256), 256, 0, stream>>>(lin1_W, Wp, 1024, 512);
    dim3 gh(Gpad / 128, 512 / 64);
    gat_gemm_wmma<<<gh, 256, (size_t)(1024 >> 5) * 4096, stream>>>(Zbf, Wp, zz, Gpad, 1024, 512);
    gat_lin_act<<<cdiv((size_t)Gpad * 512, 256), 256, 0, stream>>>(zz, lin1_b, Gpad, 512);
    gat_head<<<G_, 32, 0, stream>>>(zz, lin2_W, lin2_b, out, G_);
}
